// SelectionModel_30142080483499
// MI455X (gfx1250) — matrix operand order repaired and checked by reading, not compiled, not run on hardware
//
#include <hip/hip_runtime.h>
#include <hip/hip_bf16.h>

// ---------------------------------------------------------------------------
// MI455X / gfx1250 implementation.
//  - All GEMMs via v_wmma_f32_16x16x32_bf16 (bf16 inputs, f32 accumulate).
//  - Encoder fused (2 layers) with LDS round-trip between layers.
//  - LSTM: ONE persistent kernel (128 blocks = 1024 waves, exactly the tile
//    count). All 256 timesteps run inside the kernel with a software grid
//    barrier between steps. Cell state c lives in VGPRs for the whole
//    recurrence; W_ih/W_hh chunks are staged in LDS and shared by the 8
//    waves of a block (same hidden-tile, 8 batch-tiles).
// ---------------------------------------------------------------------------

#define INPUT_SIZE 64
#define ENC_HID    128
#define HIDDEN     512
#define GATES      2048   // 4*HIDDEN
#define NCLS       8
#define BSZ        512
#define TLEN       256
#define LSTM_BLOCKS 128   // 32 j-tiles * 4 m-groups; 8 waves each

typedef __attribute__((ext_vector_type(16))) __bf16       v16bf;
typedef __attribute__((ext_vector_type(8)))  float        v8f;
typedef __attribute__((ext_vector_type(8)))  unsigned int v8u;
typedef __attribute__((ext_vector_type(4)))  unsigned int v4u;

static_assert(sizeof(v16bf) == 32, "v16bf must be 32 bytes");

// ---- WMMA fragment loaders (contiguous k order in A and B, wave32) ----------

// A-matrix 16x32 bf16, row-major source with leading dim lda (elements).
// lanes 0-15 (g=0): V0..V7 = K 0..15 (contiguous)
// lanes16-31 (g=1): V0..V7 = K 16..31
__device__ __forceinline__ v16bf load_a_bf16(const __bf16* base, int lda,
                                             int m0, int k0, int lane) {
  int m = lane & 15, g = lane >> 4;
  const __bf16* row = base + (size_t)(m0 + m) * lda + k0;
  v4u c0 = *(const v4u*)(row + 16 * g);       // k_local {16g .. 16g+7}
  v4u c1 = *(const v4u*)(row + 16 * g + 8);   // k_local {16g+8 .. 16g+15}
  v8u r;
  r[0] = c0[0]; r[1] = c0[1]; r[2] = c0[2]; r[3] = c0[3];
  r[4] = c1[0]; r[5] = c1[1]; r[6] = c1[2]; r[7] = c1[3];
  return __builtin_bit_cast(v16bf, r);
}

// Same A layout but source is f32 (converted on the fly, used for `sequence`).
__device__ __forceinline__ v16bf load_a_f32cvt(const float* base, int lda,
                                               int m0, int k0, int lane) {
  int m = lane & 15, g = lane >> 4;
  const float* row = base + (size_t)(m0 + m) * lda + k0;
  v16bf a;
#pragma unroll
  for (int v = 0; v < 8; ++v) {
    int kl = 16 * g + 2 * v;
    a[2 * v]     = (__bf16)row[kl];
    a[2 * v + 1] = (__bf16)row[kl + 1];
  }
  return a;
}

// B-matrix 32x16 bf16 taken as W^T where W is [N,K] row-major (ldk = K):
// B(k,n) = W[n][k].  lane = n + 16*kg ; VGPR v holds (k = kg*16+2v, +1)
// -> per lane one contiguous 32-byte chunk. Works for global or LDS source.
__device__ __forceinline__ v16bf load_b_bf16(const __bf16* W, int ldk,
                                             int n0, int k0, int lane) {
  int n = lane & 15, kg = lane >> 4;
  const __bf16* p = W + (size_t)(n0 + n) * ldk + k0 + kg * 16;
  v8u r = *(const v8u*)p;
  return __builtin_bit_cast(v16bf, r);
}

__device__ __forceinline__ v8f wmma_bf16(v16bf a, v16bf b, v8f c) {
  return __builtin_amdgcn_wmma_f32_16x16x32_bf16(false, a, false, b,
                                                 (short)0, c, false, false);
}

__device__ __forceinline__ float sigmoid_(float x) {
  return 1.0f / (1.0f + __expf(-x));
}
__device__ __forceinline__ float tanh_(float x) {
  float e = __expf(-2.0f * x);
  return (1.0f - e) / (1.0f + e);
}

// ---- software grid barrier (generation counter; all blocks resident) -------
__device__ __forceinline__ void grid_sync(unsigned* cnt, unsigned* gen,
                                          unsigned nblk) {
  __threadfence();          // make this block's stores device-visible
  __syncthreads();
  if (threadIdx.x == 0) {
    unsigned g = __hip_atomic_load(gen, __ATOMIC_RELAXED,
                                   __HIP_MEMORY_SCOPE_AGENT);
    unsigned a = __hip_atomic_fetch_add(cnt, 1u, __ATOMIC_ACQ_REL,
                                        __HIP_MEMORY_SCOPE_AGENT);
    if (a == nblk - 1) {
      __hip_atomic_store(cnt, 0u, __ATOMIC_RELAXED, __HIP_MEMORY_SCOPE_AGENT);
      __hip_atomic_fetch_add(gen, 1u, __ATOMIC_RELEASE,
                             __HIP_MEMORY_SCOPE_AGENT);
    } else {
      while (__hip_atomic_load(gen, __ATOMIC_ACQUIRE,
                               __HIP_MEMORY_SCOPE_AGENT) == g) {
        __builtin_amdgcn_s_sleep(8);
      }
    }
  }
  __syncthreads();
  __threadfence();          // acquire: see other blocks' stores
}

// ---- weight preprocessing ---------------------------------------------------

__global__ void cvt_bf16_kernel(const float* __restrict__ src,
                                __bf16* __restrict__ dst, int n) {
  int i = blockIdx.x * blockDim.x + threadIdx.x;
  if (i < n) dst[i] = (__bf16)src[i];
}

__global__ void bias_comb_kernel(const float* __restrict__ a,
                                 const float* __restrict__ b,
                                 float* __restrict__ o, int n) {
  int i = blockIdx.x * blockDim.x + threadIdx.x;
  if (i < n) o[i] = a[i] + b[i];
}

// ---- fused 2-layer encoder --------------------------------------------------
// One wave handles 16 flat rows (flat = b*TLEN + t). Output is time-major
// xb[t][b][HIDDEN] in bf16 so each LSTM step reads contiguous rows.
__global__ __launch_bounds__(256) void encoder_kernel(
    const float* __restrict__ seq,   // [B*T, 64] f32
    const __bf16* __restrict__ W1b,  // [128, 64] bf16
    const float* __restrict__ b1,    // [128]
    const __bf16* __restrict__ W2b,  // [512, 128] bf16
    const float* __restrict__ b2,    // [512]
    __bf16* __restrict__ xb)         // [T, B, 512] bf16
{
  __shared__ alignas(32) __bf16 lds[8 * 16 * ENC_HID];  // 32 KB
  const int lane = threadIdx.x & 31;
  const int wave = threadIdx.x >> 5;
  const int task = blockIdx.x * 8 + wave;    // 0 .. 8191
  const int r0 = task * 16;                  // flat row base
  __bf16* myl = lds + wave * (16 * ENC_HID);

  const int n  = lane & 15;
  const int hs = lane >> 4;

  v16bf a1[2];
  a1[0] = load_a_f32cvt(seq, INPUT_SIZE, r0, 0, lane);
  a1[1] = load_a_f32cvt(seq, INPUT_SIZE, r0, 32, lane);

  // stage 1: enc1 = LeakyReLU(seq @ W1^T + b1) -> bf16 tile in LDS
  for (int t1 = 0; t1 < 8; ++t1) {
    v8f acc = {};
#pragma unroll
    for (int ks = 0; ks < 2; ++ks) {
      v16bf b = load_b_bf16(W1b, INPUT_SIZE, t1 * 16, ks * 32, lane);
      acc = wmma_bf16(a1[ks], b, acc);
    }
    float bv = b1[t1 * 16 + n];
#pragma unroll
    for (int r = 0; r < 8; ++r) {
      float x = acc[r] + bv;
      x = (x > 0.0f) ? x : 0.01f * x;                 // LeakyReLU(0.01)
      int row = r + 8 * hs;
      myl[row * ENC_HID + t1 * 16 + n] = (__bf16)x;
    }
  }

  v16bf a2[4];
#pragma unroll
  for (int ks = 0; ks < 4; ++ks)
    a2[ks] = load_a_bf16(myl, ENC_HID, 0, ks * 32, lane);

  // stage 2: x = enc1 @ W2^T + b2 -> bf16, time-major store
  for (int t2 = 0; t2 < 32; ++t2) {
    v8f acc = {};
#pragma unroll
    for (int ks = 0; ks < 4; ++ks) {
      v16bf b = load_b_bf16(W2b, ENC_HID, t2 * 16, ks * 32, lane);
      acc = wmma_bf16(a2[ks], b, acc);
    }
    float bv = b2[t2 * 16 + n];
#pragma unroll
    for (int r = 0; r < 8; ++r) {
      int fr = r0 + r + 8 * hs;                        // flat row = b*T + t
      size_t orow = (size_t)(fr & (TLEN - 1)) * BSZ + (fr >> 8);  // t*B + b
      xb[orow * HIDDEN + t2 * 16 + n] = (__bf16)(acc[r] + bv);
    }
  }
}

// ---- persistent LSTM: all 256 steps in one kernel ---------------------------
// blockIdx.x = j * 4 + mg : all 8 waves of a block share hidden-tile j and
// cover batch-tiles m = mg*8 + wave. Per 64-wide K chunk the block stages the
// W_ih and W_hh slices (4 gates x 16 cols x 64 k, bf16) into LDS once and all
// waves consume them via ds loads. Cell state c stays in VGPRs for all steps.
__global__ __launch_bounds__(256) void lstm_persistent_kernel(
    const __bf16* __restrict__ xb,    // [T, B, 512] bf16
    __bf16* hA, __bf16* hB,           // [B, 512] bf16 ping-pong (hA = zeros)
    const __bf16* __restrict__ Wih,   // [2048, 512] bf16
    const __bf16* __restrict__ Whh,   // [2048, 512] bf16
    const float* __restrict__ bias,   // [2048] = b_ih + b_hh
    float* __restrict__ h_f32,        // [B, 512] f32 final hidden state
    unsigned* ctr)                    // [2] barrier {cnt, gen}, zeroed
{
  __shared__ alignas(32) __bf16 sB[2][64 * 64];   // [ih|hh][64 rows][64 k] 16KB

  const int tid  = threadIdx.x;
  const int lane = tid & 31;
  const int wave = tid >> 5;
  const int j0 = (blockIdx.x >> 2) * 16;          // hidden-tile base
  const int m0 = ((blockIdx.x & 3) * 8 + wave) * 16;  // batch-tile base

  const int n  = lane & 15;
  const int hs = lane >> 4;
  const int col = j0 + n;

  const float bi = bias[0 * HIDDEN + col];
  const float bf = bias[1 * HIDDEN + col];
  const float bg = bias[2 * HIDDEN + col];
  const float bo = bias[3 * HIDDEN + col];

  // cooperative weight-stage indices: 4 threads per row, 32B each
  const int rr = tid >> 2;                        // 0..63 (gate*16 + n)
  const int kk = (tid & 3) * 16;                  // 0,16,32,48 (elements)
  const size_t grow = (size_t)((rr >> 4) * HIDDEN + j0 + (rr & 15)) * HIDDEN;

  float creg[8];
#pragma unroll
  for (int r = 0; r < 8; ++r) creg[r] = 0.0f;

  for (int t = 0; t < TLEN; ++t) {
    const __bf16* xt = xb + (size_t)t * BSZ * HIDDEN;
    const __bf16* hi = (t & 1) ? hB : hA;
    __bf16*       ho = (t & 1) ? hA : hB;

    v8f acc[4] = {};  // i, f, g, o

    for (int k0 = 0; k0 < HIDDEN; k0 += 64) {
      // stage W_ih / W_hh chunk for this block's j-tile
      *(v8u*)(&sB[0][rr * 64 + kk]) =
          *(const v8u*)(Wih + grow + k0 + kk);
      *(v8u*)(&sB[1][rr * 64 + kk]) =
          *(const v8u*)(Whh + grow + k0 + kk);
      __syncthreads();

#pragma unroll
      for (int ks = 0; ks < 2; ++ks) {
        v16bf ax = load_a_bf16(xt, HIDDEN, m0, k0 + ks * 32, lane);
#pragma unroll
        for (int gi = 0; gi < 4; ++gi) {
          v16bf b = load_b_bf16(sB[0], 64, gi * 16, ks * 32, lane);
          acc[gi] = wmma_bf16(ax, b, acc[gi]);
        }
        v16bf ah = load_a_bf16(hi, HIDDEN, m0, k0 + ks * 32, lane);
#pragma unroll
        for (int gi = 0; gi < 4; ++gi) {
          v16bf b = load_b_bf16(sB[1], 64, gi * 16, ks * 32, lane);
          acc[gi] = wmma_bf16(ah, b, acc[gi]);
        }
      }
      __syncthreads();   // before next chunk overwrites sB
    }

    // elementwise cell update; c stays in registers
#pragma unroll
    for (int r = 0; r < 8; ++r) {
      int row = m0 + r + 8 * hs;
      float iv = sigmoid_(acc[0][r] + bi);
      float fv = sigmoid_(acc[1][r] + bf);
      float gv = tanh_(acc[2][r] + bg);
      float ov = sigmoid_(acc[3][r] + bo);
      float cv = fv * creg[r] + iv * gv;
      creg[r] = cv;
      float hv = ov * tanh_(cv);
      size_t idx = (size_t)row * HIDDEN + col;
      ho[idx] = (__bf16)hv;
      if (t == TLEN - 1) h_f32[idx] = hv;
    }

    grid_sync(ctr, ctr + 1, LSTM_BLOCKS);   // h visible to all blocks
  }
}

// ---- final FC (512 -> 8), trivially small -----------------------------------
__global__ void fc_kernel(const float* __restrict__ h,
                          const float* __restrict__ Wfc,
                          const float* __restrict__ bfc,
                          float* __restrict__ out) {
  int idx = blockIdx.x * blockDim.x + threadIdx.x;  // 4096 threads
  if (idx >= BSZ * NCLS) return;
  int b = idx >> 3, cls = idx & 7;
  const float4* hp = (const float4*)(h + (size_t)b * HIDDEN);
  const float4* wp = (const float4*)(Wfc + (size_t)cls * HIDDEN);
  float s = 0.0f;
#pragma unroll 4
  for (int i = 0; i < HIDDEN / 4; ++i) {
    float4 x = hp[i], w = wp[i];
    s += x.x * w.x + x.y * w.y + x.z * w.z + x.w * w.w;
  }
  out[idx] = s + bfc[cls];
}

// ---------------------------------------------------------------------------

extern "C" void kernel_launch(void* const* d_in, const int* in_sizes, int n_in,
                              void* d_out, int out_size, void* d_ws,
                              size_t ws_size, hipStream_t stream) {
  const float* seq  = (const float*)d_in[0];   // [B,T,64]
  const float* W1   = (const float*)d_in[1];   // [128,64]
  const float* b1   = (const float*)d_in[2];   // [128]
  const float* W2   = (const float*)d_in[3];   // [512,128]
  const float* b2   = (const float*)d_in[4];   // [512]
  const float* Wih  = (const float*)d_in[5];   // [2048,512]
  const float* Whh  = (const float*)d_in[6];   // [2048,512]
  const float* bih  = (const float*)d_in[7];   // [2048]
  const float* bhh  = (const float*)d_in[8];   // [2048]
  const float* Wfc  = (const float*)d_in[9];   // [8,512]
  const float* bfc  = (const float*)d_in[10];  // [8]
  float* out = (float*)d_out;                  // [512,8]

  // workspace carve-out (256B aligned)
  char* ws = (char*)d_ws;
  size_t off = 0;
  auto carve = [&](size_t bytes) {
    char* p = ws + off;
    off = (off + bytes + 255) & ~(size_t)255;
    return p;
  };
  __bf16*   xb    = (__bf16*)carve((size_t)TLEN * BSZ * HIDDEN * 2);  // 128 MB
  __bf16*   W1b   = (__bf16*)carve(ENC_HID * INPUT_SIZE * 2);
  __bf16*   W2b   = (__bf16*)carve(HIDDEN * ENC_HID * 2);
  __bf16*   Wihb  = (__bf16*)carve((size_t)GATES * HIDDEN * 2);
  __bf16*   Whhb  = (__bf16*)carve((size_t)GATES * HIDDEN * 2);
  float*    biasc = (float*)carve(GATES * 4);
  __bf16*   hA    = (__bf16*)carve((size_t)BSZ * HIDDEN * 2);
  __bf16*   hB    = (__bf16*)carve((size_t)BSZ * HIDDEN * 2);
  float*    hf32  = (float*)carve((size_t)BSZ * HIDDEN * 4);
  unsigned* ctr   = (unsigned*)carve(256);
  (void)ws_size; (void)in_sizes; (void)n_in; (void)out_size;

  // 1) weight conversion f32 -> bf16 (+ combined bias)
  auto cvt = [&](const float* s, __bf16* d, int n) {
    cvt_bf16_kernel<<<(n + 255) / 256, 256, 0, stream>>>(s, d, n);
  };
  cvt(W1, W1b, ENC_HID * INPUT_SIZE);
  cvt(W2, W2b, HIDDEN * ENC_HID);
  cvt(Wih, Wihb, GATES * HIDDEN);
  cvt(Whh, Whhb, GATES * HIDDEN);
  bias_comb_kernel<<<(GATES + 255) / 256, 256, 0, stream>>>(bih, bhh, biasc,
                                                            GATES);

  // 2) zero-init recurrent state + barrier counters (every call)
  hipMemsetAsync(hA, 0, (size_t)BSZ * HIDDEN * 2, stream);
  hipMemsetAsync(ctr, 0, 256, stream);

  // 3) fused encoder: (B*T/16)/8 waves-per-block = 1024 blocks
  encoder_kernel<<<(BSZ * TLEN / 16) / 8, 256, 0, stream>>>(seq, W1b, b1, W2b,
                                                            b2, xb);

  // 4) persistent LSTM over all 256 steps (single launch)
  lstm_persistent_kernel<<<LSTM_BLOCKS, 256, 0, stream>>>(
      xb, hA, hB, Wihb, Whhb, biasc, hf32, ctr);

  // 5) classifier head
  fc_kernel<<<(BSZ * NCLS + 255) / 256, 256, 0, stream>>>(hf32, Wfc, bfc, out);
}
